// Block_new_80333068304874
// MI455X (gfx1250) — matrix operand order repaired and checked by reading, not compiled, not run on hardware
//
#include <hip/hip_runtime.h>

// ---------------------------------------------------------------------------
// SAM ViT block on gfx1250 (MI455X): wave32 WMMA f32_16x16x32_f16 everywhere,
// with async DMA (GLOBAL_LOAD_ASYNC_TO_LDS_B128) double-buffered LDS staging
// of the activation tile in all large GEMMs. Wave tile 64x64 (16 accumulators)
// for ~64 FLOP/B on the weight stream; block tile 128x256.
// Shapes: B=8, H=65 (after prompt concat), W=64, C=768, WS=14, NH=12, HD=64,
// windows Bw=200, tokens/window N=196 padded to NPAD=224, T=33280, MLP=3072.
// ---------------------------------------------------------------------------

typedef __attribute__((ext_vector_type(16))) _Float16 v16h;
typedef __attribute__((ext_vector_type(8)))  _Float16 v8h;
typedef __attribute__((ext_vector_type(8)))  float    v8f;

#define BQ     8
#define HH     65
#define WW     64
#define CC     768
#define WS     14
#define NHD    12
#define HDQ    64
#define NWIN   200          // 8 * 5 * 5
#define NPAD   224          // 196 padded to 14*16
#define MWIN   (NWIN*NPAD)  // 44800
#define TOK    33280        // 8*65*64
#define MLPD   3072

// ---------------- WMMA helpers (contiguous k = 16*(lane/16) + s, same in A and B) ---------------

__device__ __forceinline__ v8f wmma16x16x32(v16h a, v16h b, v8f c) {
  return __builtin_amdgcn_wmma_f32_16x16x32_f16(
      /*neg_a=*/false, a, /*neg_b=*/false, b,
      /*c_mod=*/(short)0, c, /*reuse_a=*/false, /*reuse_b=*/false);
}

union V16U { v16h v; v8h h[2]; };

// A matrix 16x32 (MxK), row-major source with leading dim ld (in halfs).
// lane: M = lane&15 ; lanes 0-15 hold K 0..15, lanes 16-31 hold K 16..31.
__device__ __forceinline__ v16h load_A16x32(const _Float16* p0, int ld, int lane) {
  return *(const v16h*)(p0 + (size_t)(lane & 15) * ld + ((lane >> 4) << 4));
}

// B matrix 32x16 (KxN) where source array is [N,K] row-major (B[k][n]=src[n][k]).
// lane: N = lane&15 ; lanes 0-15 hold K 0..15, lanes 16-31 hold K 16..31.
__device__ __forceinline__ v16h load_B32x16(const _Float16* p0, int ld, int lane) {
  return *(const v16h*)(p0 + (size_t)(lane & 15) * ld + ((lane >> 4) << 4));
}

// Per-lane 16B async DMA global -> LDS (tracked by ASYNCcnt).
// lds_off is the wave-relative LDS byte address (= low 32 bits of the generic
// __shared__ pointer per the ISA's flat->LDS truncation rule).
__device__ __forceinline__ void async_b128(unsigned lds_off, const void* g) {
  asm volatile("global_load_async_to_lds_b128 %0, %1, off"
               :: "v"(lds_off), "v"(g) : "memory");
}
__device__ __forceinline__ void wait_async0() {
  asm volatile("s_wait_asynccnt 0" ::: "memory");
}

// ---------------- small utility kernels ------------------------------------

__global__ void f32_to_f16_kernel(const float* __restrict__ s,
                                  _Float16* __restrict__ d, int n) {
  int i = blockIdx.x * 256 + threadIdx.x;
  if (i < n) d[i] = (_Float16)s[i];
}

// LN1 + prompt concat + window partition + pad + f32->f16.
// One block (256 thr) per windowed row; row = win*NPAD + n.
__global__ __launch_bounds__(256) void ln1_prep_kernel(
    const float* __restrict__ x, const float* __restrict__ pe,
    const float* __restrict__ g, const float* __restrict__ b,
    _Float16* __restrict__ xwin) {
  int row = blockIdx.x;
  int tid = threadIdx.x;
  int win = row / NPAD, n = row % NPAD;
  int bb = win / 25, r = win % 25;
  int h = (r / 5) * WS + n / WS;
  int w = (r % 5) * WS + n % WS;
  _Float16* dst = xwin + (size_t)row * CC;
  bool valid = (n < WS * WS) && (h < HH) && (w < WW);
  if (!valid) {
    dst[tid] = (_Float16)0.f; dst[tid + 256] = (_Float16)0.f; dst[tid + 512] = (_Float16)0.f;
    return;
  }
  const float* src = (h == 0) ? (pe + ((size_t)bb * WW + w) * CC)
                              : (x + (((size_t)bb * (HH - 1) + (h - 1)) * WW + w) * CC);
  __shared__ float red[256];
  float v0 = src[tid], v1 = src[tid + 256], v2 = src[tid + 512];
  red[tid] = v0 + v1 + v2;
  __syncthreads();
  for (int o = 128; o > 0; o >>= 1) { if (tid < o) red[tid] += red[tid + o]; __syncthreads(); }
  float mean = red[0] * (1.f / CC);
  __syncthreads();
  float d0 = v0 - mean, d1 = v1 - mean, d2 = v2 - mean;
  red[tid] = d0 * d0 + d1 * d1 + d2 * d2;
  __syncthreads();
  for (int o = 128; o > 0; o >>= 1) { if (tid < o) red[tid] += red[tid + o]; __syncthreads(); }
  float rstd = rsqrtf(red[0] * (1.f / CC) + 1e-5f);
  dst[tid]       = (_Float16)(d0 * rstd * g[tid]       + b[tid]);
  dst[tid + 256] = (_Float16)(d1 * rstd * g[tid + 256] + b[tid + 256]);
  dst[tid + 512] = (_Float16)(d2 * rstd * g[tid + 512] + b[tid + 512]);
}

// LN2 over tokens of x2 -> f16.
__global__ __launch_bounds__(256) void ln2_kernel(
    const float* __restrict__ x2, const float* __restrict__ g,
    const float* __restrict__ b, _Float16* __restrict__ h2) {
  int row = blockIdx.x;
  int tid = threadIdx.x;
  const float* src = x2 + (size_t)row * CC;
  _Float16* dst = h2 + (size_t)row * CC;
  __shared__ float red[256];
  float v0 = src[tid], v1 = src[tid + 256], v2 = src[tid + 512];
  red[tid] = v0 + v1 + v2;
  __syncthreads();
  for (int o = 128; o > 0; o >>= 1) { if (tid < o) red[tid] += red[tid + o]; __syncthreads(); }
  float mean = red[0] * (1.f / CC);
  __syncthreads();
  float d0 = v0 - mean, d1 = v1 - mean, d2 = v2 - mean;
  red[tid] = d0 * d0 + d1 * d1 + d2 * d2;
  __syncthreads();
  for (int o = 128; o > 0; o >>= 1) { if (tid < o) red[tid] += red[tid + o]; __syncthreads(); }
  float rstd = rsqrtf(red[0] * (1.f / CC) + 1e-5f);
  dst[tid]       = (_Float16)(d0 * rstd * g[tid]       + b[tid]);
  dst[tid + 256] = (_Float16)(d1 * rstd * g[tid + 256] + b[tid + 256]);
  dst[tid + 512] = (_Float16)(d2 * rstd * g[tid + 512] + b[tid + 512]);
}

// ---------------- generic WMMA GEMM: Y[M,N] = X[M,K] @ W[N,K]^T -------------
// Block: 256 thr = 8 waves as 2(M) x 4(N); wave tile 64x64; block tile 128x256.
// A (activation) tile 128x32 is DMA'd to LDS (double-buffered, ASYNCcnt), so
// the 4 N-waves share it; B (weight) fragments load direct (hot in 192MB L2).
// M % 128 == 0, N % 256 == 0, K % 32 == 0 guaranteed by problem sizes.

struct EpiQKV {
  const float* bias;
  _Float16 *Q, *K, *VT;
  __device__ __forceinline__ void store(int m, int n, float v) const {
    int win = m / NPAD, row = m % NPAD;
    float t = v + bias[n];
    int part = n / CC, rem = n % CC, head = rem >> 6, d = rem & 63;
    size_t bh = (size_t)(win * NHD + head);
    if (part == 0)      Q [bh * NPAD * HDQ + (size_t)row * HDQ + d] = (_Float16)t;
    else if (part == 1) K [bh * NPAD * HDQ + (size_t)row * HDQ + d] = (_Float16)t;
    else                VT[bh * HDQ * NPAD + (size_t)d * NPAD + row] = (_Float16)t;
  }
};

struct EpiProj {
  const float* bias;
  const float* x;
  const float* pe;
  float* x2;
  __device__ __forceinline__ void store(int m, int n, float v) const {
    int win = m / NPAD, t = m % NPAD;
    if (t >= WS * WS) return;
    int bb = win / 25, r = win % 25;
    int h = (r / 5) * WS + t / WS;
    int w = (r % 5) * WS + t % WS;
    if (h >= HH || w >= WW) return;
    float sc = (h == 0) ? pe[((size_t)bb * WW + w) * CC + n]
                        : x[(((size_t)bb * (HH - 1) + (h - 1)) * WW + w) * CC + n];
    x2[(((size_t)bb * HH + h) * WW + w) * CC + n] = v + bias[n] + sc;
  }
};

struct EpiMlp1 {
  const float* bias;
  _Float16* out;
  __device__ __forceinline__ void store(int m, int n, float v) const {
    float t = v + bias[n];
    float gel = 0.5f * t * (1.f + erff(t * 0.70710678118654752f));
    out[(size_t)m * MLPD + n] = (_Float16)gel;
  }
};

struct EpiMlp2 {
  const float* bias;
  const float* x2;
  float* out;
  __device__ __forceinline__ void store(int m, int n, float v) const {
    size_t i = (size_t)m * CC + n;
    out[i] = v + bias[n] + x2[i];
  }
};

template <typename Epi>
__global__ __launch_bounds__(256) void wmma_gemm_kernel(
    const _Float16* __restrict__ X, int ldx,
    const _Float16* __restrict__ W, int ldw,
    int K, Epi epi) {
  __shared__ __align__(32) _Float16 stage[2][128 * 32];   // 2 x 8KB A tiles
  int tid = threadIdx.x;
  int lane = tid & 31, wave = tid >> 5;
  int wm = wave >> 2, wn = wave & 3;
  size_t mblk = (size_t)blockIdx.y * 128;
  size_t n0 = (size_t)blockIdx.x * 256 + (size_t)wn * 64;

  // per-thread DMA slice: two B128s (rows srow and srow+64) of the 128x32 tile
  int srow = tid >> 2;              // 0..63
  int sseg = (tid & 3) << 3;        // 0,8,16,24 (halfs)
  const _Float16* g0 = X + (mblk + srow) * (size_t)ldx + sseg;
  const _Float16* g1 = X + (mblk + srow + 64) * (size_t)ldx + sseg;
  unsigned l00 = (unsigned)(uintptr_t)&stage[0][srow * 32 + sseg];
  unsigned l01 = (unsigned)(uintptr_t)&stage[0][(srow + 64) * 32 + sseg];
  unsigned l10 = (unsigned)(uintptr_t)&stage[1][srow * 32 + sseg];
  unsigned l11 = (unsigned)(uintptr_t)&stage[1][(srow + 64) * 32 + sseg];

  v8f acc[4][4] = {};
  const _Float16* Wp = W + n0 * (size_t)ldw;
  const _Float16* Ab0 = &stage[0][(wm * 64) * 32];
  const _Float16* Ab1 = &stage[1][(wm * 64) * 32];

  async_b128(l00, g0);                            // prologue: k = 0 -> buf0
  async_b128(l01, g1);
  int nk = K >> 5;
  for (int it = 0; it < nk; ++it) {
    wait_async0();                                // own DMA for buf[it&1] done
    __syncthreads();                              // everyone's DMA visible
    if (it + 1 < nk) {                            // overlap next DMA w/ compute
      const _Float16* gk0 = g0 + (size_t)(it + 1) * 32;
      const _Float16* gk1 = g1 + (size_t)(it + 1) * 32;
      if (it & 1) { async_b128(l00, gk0); async_b128(l01, gk1); }
      else        { async_b128(l10, gk0); async_b128(l11, gk1); }
    }
    const _Float16* Ab = (it & 1) ? Ab1 : Ab0;
    int k = it << 5;
    v16h a[4];
#pragma unroll
    for (int mt = 0; mt < 4; ++mt)
      a[mt] = load_A16x32(Ab + mt * 16 * 32, 32, lane);
#pragma unroll
    for (int nt = 0; nt < 4; ++nt) {
      v16h bfr = load_B32x16(Wp + (size_t)nt * 16 * ldw + k, ldw, lane);
#pragma unroll
      for (int mt = 0; mt < 4; ++mt)
        acc[mt][nt] = wmma16x16x32(a[mt], bfr, acc[mt][nt]);
    }
  }
  int nl = lane & 15, hi = lane >> 4;
  size_t m0 = mblk + (size_t)wm * 64;
#pragma unroll
  for (int mt = 0; mt < 4; ++mt)
#pragma unroll
    for (int nt = 0; nt < 4; ++nt)
#pragma unroll
      for (int r = 0; r < 8; ++r)
        epi.store((int)m0 + mt * 16 + r + hi * 8,
                  (int)n0 + nt * 16 + nl, acc[mt][nt][r]);
}

// ---------------- attention: one block (128 thr = 4 waves) per (win, head) --

__global__ __launch_bounds__(128) void attn_kernel(
    const _Float16* __restrict__ Q, const _Float16* __restrict__ Kb,
    const _Float16* __restrict__ VT,
    const float* __restrict__ rph, const float* __restrict__ rpw,
    _Float16* __restrict__ aout) {
  int bh = blockIdx.x;                 // win*NHD + head
  int win = bh / NHD, head = bh % NHD;
  const _Float16* Qh = Q  + (size_t)bh * NPAD * HDQ;
  const _Float16* Kh = Kb + (size_t)bh * NPAD * HDQ;
  const _Float16* Vh = VT + (size_t)bh * HDQ * NPAD;

  __shared__ float relH[WS * WS * WS];        // [q (=qh*14+qw)][kh]
  __shared__ float relW[WS * WS * WS];        // [q][kw]
  __shared__ __align__(32) _Float16 Pbuf[4][16 * NPAD];

  int tid = threadIdx.x;
  // rel_h[q][k] = sum_c Q[q][c] * rel_pos_h[qh-k+13][c]   (unscaled q, SAM-style)
  for (int i = tid; i < WS * WS * WS; i += 128) {
    int qh = i / (WS * WS), rem = i % (WS * WS), qw = rem / WS, kk = rem % WS;
    const float* rh = rph + (size_t)(qh - kk + WS - 1) * HDQ;
    const float* rw = rpw + (size_t)(qw - kk + WS - 1) * HDQ;
    const _Float16* qp = Qh + (size_t)(qh * WS + qw) * HDQ;
    float sh = 0.f, sw = 0.f;
#pragma unroll 8
    for (int c = 0; c < HDQ; ++c) {
      float qc = (float)qp[c];
      sh += qc * rh[c];
      sw += qc * rw[c];
    }
    relH[i] = sh;
    relW[i] = sw;
  }
  __syncthreads();

  int lane = tid & 31, wave = tid >> 5;
  int nl = lane & 15, hi = lane >> 4;
  size_t obase = ((size_t)win * NPAD) * CC + (size_t)head * HDQ;

  for (int qt = wave; qt < NPAD / 16; qt += 4) {
    // ---- S = (Q K^T) * 1/8 + rel, over all 224 (padded) keys ----
    v8f S[14] = {};
    for (int kk = 0; kk < HDQ; kk += 32) {
      v16h a = load_A16x32(Qh + (size_t)qt * 16 * HDQ + kk, HDQ, lane);
#pragma unroll
      for (int jt = 0; jt < 14; ++jt) {
        v16h bfr = load_B32x16(Kh + (size_t)jt * 16 * HDQ + kk, HDQ, lane);
        S[jt] = wmma16x16x32(a, bfr, S[jt]);
      }
    }
    // ---- scale + rel-pos + mask + row softmax (rows live in 16-lane halves) ----
#pragma unroll
    for (int r = 0; r < 8; ++r) {
      int q = qt * 16 + r + hi * 8;
      float mx = -3.0e38f;
#pragma unroll
      for (int jt = 0; jt < 14; ++jt) {
        int j = jt * 16 + nl;
        float v = S[jt][r] * 0.125f;
        if (j < WS * WS && q < WS * WS)
          v += relH[q * WS + j / WS] + relW[q * WS + j % WS];
        if (j >= WS * WS) v = -1.0e30f;
        S[jt][r] = v;
        mx = fmaxf(mx, v);
      }
      for (int o = 1; o < 16; o <<= 1) mx = fmaxf(mx, __shfl_xor(mx, o, 32));
      float sum = 0.f;
#pragma unroll
      for (int jt = 0; jt < 14; ++jt) {
        float p = __expf(S[jt][r] - mx);
        S[jt][r] = p;
        sum += p;
      }
      for (int o = 1; o < 16; o <<= 1) sum += __shfl_xor(sum, o, 32);
      float inv = 1.f / sum;
#pragma unroll
      for (int jt = 0; jt < 14; ++jt)
        Pbuf[wave][(r + hi * 8) * NPAD + jt * 16 + nl] = (_Float16)(S[jt][r] * inv);
    }
    // same-wave LDS store -> load ordering (waves own disjoint Pbuf regions)
    asm volatile("s_wait_dscnt 0" ::: "memory");

    // ---- O = P @ V  (A from LDS, B from V^T [d, token]) ----
    v8f O[4] = {};
    const _Float16* Pl = &Pbuf[wave][0];
    for (int cc = 0; cc < NPAD / 32; ++cc) {
      v16h a = load_A16x32(Pl + cc * 32, NPAD, lane);
#pragma unroll
      for (int dt = 0; dt < 4; ++dt) {
        v16h bfr = load_B32x16(Vh + (size_t)dt * 16 * NPAD + cc * 32, NPAD, lane);
        O[dt] = wmma16x16x32(a, bfr, O[dt]);
      }
    }
#pragma unroll
    for (int dt = 0; dt < 4; ++dt)
#pragma unroll
      for (int r = 0; r < 8; ++r) {
        int q = qt * 16 + r + hi * 8;
        aout[obase + (size_t)q * CC + dt * 16 + nl] = (_Float16)O[dt][r];
      }
  }
}

// ---------------------------------------------------------------------------

extern "C" void kernel_launch(void* const* d_in, const int* in_sizes, int n_in,
                              void* d_out, int out_size, void* d_ws, size_t ws_size,
                              hipStream_t stream) {
  (void)in_sizes; (void)n_in; (void)out_size; (void)ws_size;
  const float* x     = (const float*)d_in[0];
  const float* pe    = (const float*)d_in[1];
  const float* ln1g  = (const float*)d_in[2];
  const float* ln1b  = (const float*)d_in[3];
  const float* qkvw  = (const float*)d_in[4];
  const float* qkvb  = (const float*)d_in[5];
  const float* rph   = (const float*)d_in[6];
  const float* rpw   = (const float*)d_in[7];
  const float* projw = (const float*)d_in[8];
  const float* projb = (const float*)d_in[9];
  const float* ln2g  = (const float*)d_in[10];
  const float* ln2b  = (const float*)d_in[11];
  const float* w1    = (const float*)d_in[12];
  const float* b1    = (const float*)d_in[13];
  const float* w2    = (const float*)d_in[14];
  const float* b2    = (const float*)d_in[15];
  float* out = (float*)d_out;

  char* ws = (char*)d_ws;
  size_t off = 0;
  auto take = [&](size_t bytes) -> char* {
    char* p = ws + off;
    off = (off + bytes + 255) & ~(size_t)255;
    return p;
  };
  _Float16* wqkv = (_Float16*)take((size_t)3 * CC * CC * 2);
  _Float16* wprj = (_Float16*)take((size_t)CC * CC * 2);
  _Float16* wm1  = (_Float16*)take((size_t)MLPD * CC * 2);
  _Float16* wm2  = (_Float16*)take((size_t)CC * MLPD * 2);
  float*    x2   = (float*)   take((size_t)TOK * CC * 4);
  _Float16* xwin = (_Float16*)take((size_t)MWIN * CC * 2);
  _Float16* Qb   = (_Float16*)take((size_t)NWIN * NHD * NPAD * HDQ * 2);
  _Float16* Kbf  = (_Float16*)take((size_t)NWIN * NHD * NPAD * HDQ * 2);
  _Float16* VTb  = (_Float16*)take((size_t)NWIN * NHD * HDQ * NPAD * 2);
  // Aliases (lifetimes are disjoint):
  _Float16* aout = xwin;   // attention out reuses xwin (dead after qkv GEMM)
  _Float16* h2   = VTb;    // LN2 out reuses V^T (dead after attention)
  _Float16* m1   = xwin;   // GELU out spans xwin+Q+K (206 MB >= 204 MB needed)

  // weight f32 -> f16 (layout [N,K] is exactly the B-fragment layout)
  {
    int n;
    n = 3 * CC * CC;  f32_to_f16_kernel<<<(n + 255) / 256, 256, 0, stream>>>(qkvw,  wqkv, n);
    n = CC * CC;      f32_to_f16_kernel<<<(n + 255) / 256, 256, 0, stream>>>(projw, wprj, n);
    n = MLPD * CC;    f32_to_f16_kernel<<<(n + 255) / 256, 256, 0, stream>>>(w1,    wm1,  n);
    n = CC * MLPD;    f32_to_f16_kernel<<<(n + 255) / 256, 256, 0, stream>>>(w2,    wm2,  n);
  }

  // LN1 + concat + window partition (pads zeroed)
  ln1_prep_kernel<<<MWIN, 256, 0, stream>>>(x, pe, ln1g, ln1b, xwin);

  // QKV: [44800,768] @ [2304,768]^T -> scatter Q / K / V^T per window-head
  wmma_gemm_kernel<EpiQKV><<<dim3((3 * CC) / 256, MWIN / 128), 256, 0, stream>>>(
      xwin, CC, wqkv, CC, CC, EpiQKV{qkvb, Qb, Kbf, VTb});

  // attention per (window, head)
  attn_kernel<<<NWIN * NHD, 128, 0, stream>>>(Qb, Kbf, VTb, rph, rpw, aout);

  // proj + unpartition + crop + residual -> x2 (f32)
  wmma_gemm_kernel<EpiProj><<<dim3(CC / 256, MWIN / 128), 256, 0, stream>>>(
      aout, CC, wprj, CC, CC, EpiProj{projb, x, pe, x2});

  // LN2
  ln2_kernel<<<TOK, 256, 0, stream>>>(x2, ln2g, ln2b, h2);

  // MLP1 + GELU
  wmma_gemm_kernel<EpiMlp1><<<dim3(MLPD / 256, TOK / 128), 256, 0, stream>>>(
      h2, CC, wm1, CC, CC, EpiMlp1{b1, m1});

  // MLP2 + residual -> out
  wmma_gemm_kernel<EpiMlp2><<<dim3(CC / 256, TOK / 128), 256, 0, stream>>>(
      m1, MLPD, wm2, MLPD, MLPD, EpiMlp2{b2, x2, out});
}
